// enc_mtan_rnn_25091198943725
// MI455X (gfx1250) — matrix operand order repaired and checked by reading, not compiled, not run on hardware
//
#include <hip/hip_runtime.h>
#include <math.h>

typedef __attribute__((ext_vector_type(16))) _Float16 v16h;
typedef __attribute__((ext_vector_type(8)))  _Float16 v8h;
typedef __attribute__((ext_vector_type(8)))  float    v8f;

// Problem constants
#define Bb_  16
#define LK_  256
#define LQ_  128
#define Dd_  32
#define Ee_  128
#define Hh_  128

static __device__ __forceinline__ v16h frag_combine(v8h lo, v8h hi) {
  v16h a;
#pragma unroll
  for (int i = 0; i < 8; ++i) { a[i] = lo[i]; a[8 + i] = hi[i]; }
  return a;
}

// ---------------------------------------------------------------------------
// Generic batched WMMA GEMM:  C[b] = act( scale * A[b] @ Bt[b]^T + bias )
// A   : (M x K) f16 row-major          (A-fragment: 2x b128 per lane)
// Bt  : (N x K) f16 row-major (i.e. B col-major; B-fragment: 2x b128 per lane)
// out : f32 (Cf) or f16 (Ch), ldc row stride, only n < Nvalid written
// Each wave computes one 16x16 tile; block = 4 waves.
// ---------------------------------------------------------------------------
__global__ void gemm_wmma_kernel(const _Float16* __restrict__ A, int lda, long sA,
                                 const _Float16* __restrict__ Bt, int ldb, long sB,
                                 float* __restrict__ Cf, _Float16* __restrict__ Ch,
                                 int ldc, long sC,
                                 const float* __restrict__ bias,
                                 int N, int K, int Nvalid, float scale, int relu)
{
  const int wave = threadIdx.x >> 5;
  const int lane = threadIdx.x & 31;
  const int tn = blockIdx.x * 4 + wave;   // N tile (wave-uniform)
  const int tm = blockIdx.y;              // M tile
  const int b  = blockIdx.z;
  if (tn * 16 >= N) return;               // whole-wave exit, EXEC stays all-1s

  // 16-bit A 16x32 fragment: lane m = lane&15; K halves contiguous (koff+0..15)
  const _Float16* Ar = A + (long)b * sA + (long)(tm * 16 + (lane & 15)) * lda
                         + ((lane >> 4) * 16);
  // 16-bit B 32x16 fragment via col-major: lane n = lane&15; K halves contiguous
  const _Float16* Br = Bt + (long)b * sB + (long)(tn * 16 + (lane & 15)) * ldb
                          + ((lane >> 4) * 16);

  v8f acc = {};
  for (int k0 = 0; k0 < K; k0 += 32) {
    v8h alo = *reinterpret_cast<const v8h*>(Ar + k0);
    v8h ahi = *reinterpret_cast<const v8h*>(Ar + k0 + 8);
    v16h af = frag_combine(alo, ahi);
    v16h bf = *reinterpret_cast<const v16h*>(Br + k0);
    acc = __builtin_amdgcn_wmma_f32_16x16x32_f16(false, af, false, bf,
                                                 (short)0, acc, false, false);
  }

  const int nl = lane & 15;
  const int ml = (lane >> 4) * 8;          // C/D layout: lanes>=16 hold M=+8
  const int n  = tn * 16 + nl;
  if (n >= Nvalid) return;                 // post-WMMA divergence is fine
  const float bv = bias ? bias[n] : 0.0f;
#pragma unroll
  for (int i = 0; i < 8; ++i) {
    const int m = tm * 16 + ml + i;
    float v = acc[i] * scale + bv;
    if (relu) v = fmaxf(v, 0.0f);
    const long o = (long)b * sC + (long)m * ldc + n;
    if (Cf) Cf[o] = v;
    else    Ch[o] = (_Float16)v;
  }
}

// ---------------------------------------------------------------------------
// GRU recurrence (one direction). One workgroup, 8 waves.
// gh = h @ Whh + bhh done as WMMA: M=16(batch), N=384, K=128, h f16 in LDS.
// Wave w owns N-tiles {w, w+8, w+16} -> matching r/z/n gate columns 16w..16w+15.
// gx (precomputed x@Wih + bih) read from global, f32.
// ---------------------------------------------------------------------------
__global__ void gru_wmma_kernel(const float* __restrict__ gx,      // (B*LQ x 384)
                                const _Float16* __restrict__ whhT, // (384 x 128)
                                const float* __restrict__ gbhh,    // (384)
                                _Float16* __restrict__ yout,       // (B*LQ x 256)
                                int yoff, int reverse)
{
  __shared__ __align__(16) _Float16 hsh[16 * 128];  // h, f16 copy for WMMA A
  __shared__ float hss[16 * 128];                   // h, f32 state
  const int wave = threadIdx.x >> 5;
  const int lane = threadIdx.x & 31;

  for (int i = threadIdx.x; i < 16 * 128; i += 256) {
    hss[i] = 0.0f; hsh[i] = (_Float16)0.0f;
  }
  __syncthreads();

  const int am  = lane & 15;              // A fragment row = batch
  const int ak  = (lane >> 4) * 16;       // A fragment K offset
  const int nl  = lane & 15;
  const int ml  = (lane >> 4) * 8;
  const int col = wave * 16 + nl;         // hidden column owned by this lane
  const _Float16* wr = whhT + (long)(      wave * 16 + nl) * 128 + ((lane >> 4) * 16);
  const _Float16* wz = whhT + (long)(128 + wave * 16 + nl) * 128 + ((lane >> 4) * 16);
  const _Float16* wn = whhT + (long)(256 + wave * 16 + nl) * 128 + ((lane >> 4) * 16);

  for (int step = 0; step < LQ_; ++step) {
    const int t = reverse ? (LQ_ - 1 - step) : step;
    v8f aR = {}, aZ = {}, aN = {};
#pragma unroll
    for (int k0 = 0; k0 < 128; k0 += 32) {
      v8h alo = *reinterpret_cast<const v8h*>(hsh + am * 128 + k0 + ak);
      v8h ahi = *reinterpret_cast<const v8h*>(hsh + am * 128 + k0 + ak + 8);
      v16h af = frag_combine(alo, ahi);
      v16h bR = *reinterpret_cast<const v16h*>(wr + k0);
      v16h bZ = *reinterpret_cast<const v16h*>(wz + k0);
      v16h bN = *reinterpret_cast<const v16h*>(wn + k0);
      aR = __builtin_amdgcn_wmma_f32_16x16x32_f16(false, af, false, bR, (short)0, aR, false, false);
      aZ = __builtin_amdgcn_wmma_f32_16x16x32_f16(false, af, false, bZ, (short)0, aZ, false, false);
      aN = __builtin_amdgcn_wmma_f32_16x16x32_f16(false, af, false, bN, (short)0, aN, false, false);
    }
    float hnew[8];
#pragma unroll
    for (int i = 0; i < 8; ++i) {
      const int m = ml + i;                              // batch index
      const long gr = ((long)m * LQ_ + t) * 384;
      const float xr = gx[gr + col];
      const float xz = gx[gr + 128 + col];
      const float xn = gx[gr + 256 + col];
      const float r = 1.0f / (1.0f + expf(-(xr + aR[i] + gbhh[col])));
      const float z = 1.0f / (1.0f + expf(-(xz + aZ[i] + gbhh[128 + col])));
      const float n = tanhf(xn + r * (aN[i] + gbhh[256 + col]));
      const float hold = hss[m * 128 + col];
      hnew[i] = (1.0f - z) * n + z * hold;
      yout[((long)m * LQ_ + t) * 256 + yoff + col] = (_Float16)hnew[i];
    }
    __syncthreads();   // all reads of h done
#pragma unroll
    for (int i = 0; i < 8; ++i) {
      const int m = ml + i;
      hss[m * 128 + col] = hnew[i];
      hsh[m * 128 + col] = (_Float16)hnew[i];
    }
    __syncthreads();   // h updated for next step
  }
}

// ---------------------------------------------------------------------------
// Prep / elementwise kernels
// ---------------------------------------------------------------------------

// dst (Cpad x Rpad, f16) = transpose(src (R x C, f32)), zero padded
__global__ void wT_kernel(const float* __restrict__ src, int R, int C,
                          _Float16* __restrict__ dst, int Cpad, int Rpad)
{
  const long idx = (long)blockIdx.x * 256 + threadIdx.x;
  const long tot = (long)Cpad * Rpad;
  if (idx >= tot) return;
  const int  r = (int)(idx % Rpad);
  const long c = idx / Rpad;
  const float v = (c < C && r < R) ? src[(long)r * C + c] : 0.0f;
  dst[idx] = (_Float16)v;
}

__global__ void biaspad_kernel(const float* __restrict__ src, int n,
                               float* __restrict__ dst, int npad)
{
  const int i = blockIdx.x * 256 + threadIdx.x;
  if (i < npad) dst[i] = (i < n) ? src[i] : 0.0f;
}

// time embedding for keys (rows 0..4095 -> time_steps) and queries (rows 4096..4223)
__global__ void embed_kernel(const float* __restrict__ time_steps,
                             const float* __restrict__ query,
                             const float* __restrict__ lin_w, const float* __restrict__ lin_b,
                             const float* __restrict__ per_w, const float* __restrict__ per_b,
                             _Float16* __restrict__ key_e, _Float16* __restrict__ q_e)
{
  const long idx = (long)blockIdx.x * 256 + threadIdx.x;
  if (idx >= (long)(Bb_ * LK_ + LQ_) * Ee_) return;
  const int col = (int)(idx & (Ee_ - 1));
  const long row = idx >> 7;
  const float t = (row < Bb_ * LK_) ? time_steps[row] : query[row - Bb_ * LK_];
  const float v = (col == 0) ? (t * lin_w[0] + lin_b[0])
                             : sinf(t * per_w[col - 1] + per_b[col - 1]);
  if (row < Bb_ * LK_) key_e[row * Ee_ + col] = (_Float16)v;
  else                 q_e[(row - Bb_ * LK_) * Ee_ + col] = (_Float16)v;
}

// xT[b] (64 x 256) f16 = transpose of x[b] (256 x 64) f32
__global__ void xT_kernel(const float* __restrict__ x, _Float16* __restrict__ xT)
{
  const long idx = (long)blockIdx.x * 256 + threadIdx.x;
  if (idx >= (long)Bb_ * LK_ * 64) return;
  const int c = (int)(idx & 63);
  const long bl = idx >> 6;
  const int l = (int)(bl & (LK_ - 1));
  const int b = (int)(bl >> 8);
  xT[((long)b * 64 + c) * LK_ + l] = (_Float16)x[idx];
}

// per-row (256 wide) max + exp; scores f32 in, e f16 out
__global__ void rowmax_exp_kernel(const float* __restrict__ s, _Float16* __restrict__ eh)
{
  __shared__ float red[256];
  const long row = blockIdx.x;                 // 0..2047 = b*LQ+q
  const float v = s[row * 256 + threadIdx.x];
  red[threadIdx.x] = v;
  __syncthreads();
  for (int off = 128; off > 0; off >>= 1) {
    if (threadIdx.x < off) red[threadIdx.x] = fmaxf(red[threadIdx.x], red[threadIdx.x + off]);
    __syncthreads();
  }
  eh[row * 256 + threadIdx.x] = (_Float16)expf(v - red[0]);
}

// att = num / denom   (denom = mask columns 32..63 of num); f16 out for next GEMM
__global__ void att_div_kernel(const float* __restrict__ num, _Float16* __restrict__ atth)
{
  const long idx = (long)blockIdx.x * 256 + threadIdx.x;
  if (idx >= (long)Bb_ * LQ_ * 64) return;
  const int c = (int)(idx & 63);
  const long row = idx >> 6;
  const float den = num[row * 64 + 32 + (c & 31)];
  atth[idx] = (_Float16)(den > 0.0f ? num[idx] / den : 0.0f);
}

// ---------------------------------------------------------------------------
extern "C" void kernel_launch(void* const* d_in, const int* in_sizes, int n_in,
                              void* d_out, int out_size, void* d_ws, size_t ws_size,
                              hipStream_t stream)
{
  (void)in_sizes; (void)n_in; (void)out_size; (void)ws_size;
  const float* x      = (const float*)d_in[0];
  const float* tsteps = (const float*)d_in[1];
  const float* query  = (const float*)d_in[2];
  const float* wq     = (const float*)d_in[3];
  const float* bq     = (const float*)d_in[4];
  const float* wk     = (const float*)d_in[5];
  const float* bk     = (const float*)d_in[6];
  const float* wo     = (const float*)d_in[7];
  const float* bo     = (const float*)d_in[8];
  const float* per_w  = (const float*)d_in[9];
  const float* per_b  = (const float*)d_in[10];
  const float* lin_w  = (const float*)d_in[11];
  const float* lin_b  = (const float*)d_in[12];
  const float* gwih_f = (const float*)d_in[13];
  const float* gwhh_f = (const float*)d_in[14];
  const float* gbih_f = (const float*)d_in[15];
  const float* gbhh_f = (const float*)d_in[16];
  const float* gwih_b = (const float*)d_in[17];
  const float* gwhh_b = (const float*)d_in[18];
  const float* gbih_b = (const float*)d_in[19];
  const float* gbhh_b = (const float*)d_in[20];
  const float* mw1    = (const float*)d_in[21];
  const float* mb1    = (const float*)d_in[22];
  const float* mw2    = (const float*)d_in[23];
  const float* mb2    = (const float*)d_in[24];
  float* out = (float*)d_out;

  // workspace carve-up (256B aligned)
  char* wsb = (char*)d_ws;
  size_t off = 0;
  auto alloc = [&](size_t bytes) -> char* {
    char* p = wsb + off;
    off = (off + bytes + 255) & ~(size_t)255;
    return p;
  };
  _Float16* wqT    = (_Float16*)alloc(128 * 128 * 2);
  _Float16* wkT    = (_Float16*)alloc(128 * 128 * 2);
  _Float16* woT    = (_Float16*)alloc(128 * 64 * 2);
  _Float16* wihfT  = (_Float16*)alloc(384 * 128 * 2);
  _Float16* whhfT  = (_Float16*)alloc(384 * 128 * 2);
  _Float16* wihbT  = (_Float16*)alloc(384 * 128 * 2);
  _Float16* whhbT  = (_Float16*)alloc(384 * 128 * 2);
  _Float16* mw1T   = (_Float16*)alloc(64 * 256 * 2);
  _Float16* mw2T   = (_Float16*)alloc(48 * 64 * 2);
  float*    mb1p   = (float*)alloc(64 * 4);
  float*    mb2p   = (float*)alloc(48 * 4);
  _Float16* keyEh  = (_Float16*)alloc((size_t)4096 * 128 * 2);
  _Float16* qEh    = (_Float16*)alloc(128 * 128 * 2);
  _Float16* kmatH  = (_Float16*)alloc((size_t)4096 * 128 * 2);
  _Float16* qmatH  = (_Float16*)alloc(128 * 128 * 2);
  _Float16* xTh    = (_Float16*)alloc((size_t)16 * 64 * 256 * 2);
  float*    sbuf   = (float*)alloc((size_t)2048 * 256 * 4);
  _Float16* ehbuf  = (_Float16*)alloc((size_t)2048 * 256 * 2);
  float*    numb   = (float*)alloc((size_t)2048 * 64 * 4);
  _Float16* attH   = (_Float16*)alloc((size_t)2048 * 64 * 2);
  _Float16* hiddH  = (_Float16*)alloc((size_t)2048 * 128 * 2);
  float*    gxf    = (float*)alloc((size_t)2048 * 384 * 4);
  float*    gxb    = (float*)alloc((size_t)2048 * 384 * 4);
  _Float16* yh     = (_Float16*)alloc((size_t)2048 * 256 * 2);
  _Float16* z1h    = (_Float16*)alloc((size_t)2048 * 64 * 2);

  auto wT = [&](const float* src, int R, int C, _Float16* dst, int Cpad, int Rpad) {
    const long tot = (long)Cpad * Rpad;
    wT_kernel<<<(unsigned)((tot + 255) / 256), 256, 0, stream>>>(src, R, C, dst, Cpad, Rpad);
  };
  auto gemm = [&](const _Float16* A, int lda, long sA,
                  const _Float16* Bt, int ldb, long sB,
                  float* Cf, _Float16* Ch, int ldc, long sC, const float* bias,
                  int M, int N, int K, int Nvalid, float scale, int relu, int batches) {
    dim3 grid((unsigned)((N / 16 + 3) / 4), (unsigned)(M / 16), (unsigned)batches);
    gemm_wmma_kernel<<<grid, 128, 0, stream>>>(A, lda, sA, Bt, ldb, sB,
                                               Cf, Ch, ldc, sC, bias, N, K, Nvalid, scale, relu);
  };

  // 0) weight transpose/convert + bias pads
  wT(wq, 128, 128, wqT, 128, 128);
  wT(wk, 128, 128, wkT, 128, 128);
  wT(wo, 64, 128, woT, 128, 64);
  wT(gwih_f, 128, 384, wihfT, 384, 128);
  wT(gwhh_f, 128, 384, whhfT, 384, 128);
  wT(gwih_b, 128, 384, wihbT, 384, 128);
  wT(gwhh_b, 128, 384, whhbT, 384, 128);
  wT(mw1, 256, 50, mw1T, 64, 256);
  wT(mw2, 50, 40, mw2T, 48, 64);
  biaspad_kernel<<<1, 256, 0, stream>>>(mb1, 50, mb1p, 64);
  biaspad_kernel<<<1, 256, 0, stream>>>(mb2, 40, mb2p, 48);

  // 1) time embeddings (key + query), f16
  embed_kernel<<<(unsigned)(((long)(4096 + 128) * 128 + 255) / 256), 256, 0, stream>>>(
      tsteps, query, lin_w, lin_b, per_w, per_b, keyEh, qEh);
  // x transposed per batch for attention GEMM B-operand
  xT_kernel<<<(unsigned)(((long)16 * 256 * 64 + 255) / 256), 256, 0, stream>>>(x, xTh);

  // 2) projections: qmat = q_e@wq+bq ; kmat = key_e@wk+bk  (f16 out)
  gemm(qEh, 128, 0, wqT, 128, 0, nullptr, qmatH, 128, 0, bq, 128, 128, 128, 128, 1.0f, 0, 1);
  gemm(keyEh, 128, 0, wkT, 128, 0, nullptr, kmatH, 128, 0, bk, 4096, 128, 128, 128, 1.0f, 0, 1);

  // 3) scores[b] = qmat @ kmat[b]^T / sqrt(128)   (f32)
  const float invsqrt = 1.0f / sqrtf(128.0f);
  gemm(qmatH, 128, 0, kmatH, 128, (long)256 * 128,
       sbuf, nullptr, 256, (long)128 * 256, nullptr, 128, 256, 128, 256, invsqrt, 0, 16);

  // 4) row softmax numerators: e = exp(s - rowmax), f16
  rowmax_exp_kernel<<<2048, 256, 0, stream>>>(sbuf, ehbuf);

  // 5) num[b] = e[b] @ x[b]  (cols 32..63 are the per-channel denominators)
  gemm(ehbuf, 256, (long)128 * 256, xTh, 256, (long)64 * 256,
       numb, nullptr, 64, (long)128 * 64, nullptr, 128, 64, 256, 64, 1.0f, 0, 16);

  // 6) att = num/denom (f16)
  att_div_kernel<<<(unsigned)(((long)2048 * 64 + 255) / 256), 256, 0, stream>>>(numb, attH);

  // 7) hidd = att @ wo + bo  (f16)
  gemm(attH, 64, 0, woT, 64, 0, nullptr, hiddH, 128, 0, bo, 2048, 128, 64, 128, 1.0f, 0, 1);

  // 8) GRU input gates for all timesteps at once: gx = hidd @ wih + bih (f32)
  gemm(hiddH, 128, 0, wihfT, 128, 0, gxf, nullptr, 384, 0, gbih_f, 2048, 384, 128, 384, 1.0f, 0, 1);
  gemm(hiddH, 128, 0, wihbT, 128, 0, gxb, nullptr, 384, 0, gbih_b, 2048, 384, 128, 384, 1.0f, 0, 1);

  // 9) GRU recurrences (serial over 128 steps, LDS-resident h)
  gru_wmma_kernel<<<1, 256, 0, stream>>>(gxf, whhfT, gbhh_f, yh, 0, 0);
  gru_wmma_kernel<<<1, 256, 0, stream>>>(gxb, whhbT, gbhh_b, yh, 128, 1);

  // 10) MLP: z1 = relu(y @ mw1 + mb1) (f16, padded cols are exactly 0)
  gemm(yh, 256, 0, mw1T, 256, 0, nullptr, z1h, 64, 0, mb1p, 2048, 64, 256, 64, 1.0f, 1, 1);
  // out = z1 @ mw2 + mb2, write only valid 40 columns straight to d_out
  gemm(z1h, 64, 0, mw2T, 64, 0, out, nullptr, 40, 0, mb2p, 2048, 48, 64, 40, 1.0f, 0, 1);
}
